// WindowedAttentionBlock_70738111365323
// MI455X (gfx1250) — compile-verified, matrix operand order checked by reading, not run on hardware
//
#include <hip/hip_runtime.h>
#include <hip/hip_bf16.h>
#include <math.h>
#include <stdint.h>

// ---------- types ----------
typedef __bf16 bf16_t;
typedef __bf16 v16bf __attribute__((ext_vector_type(16)));
typedef float  v8f   __attribute__((ext_vector_type(8)));
typedef unsigned int u32x4 __attribute__((ext_vector_type(4)));
typedef int i32x4 __attribute__((ext_vector_type(4)));
typedef int i32x8 __attribute__((ext_vector_type(8)));

#define TS    16384            // tokens per sample (128*128)
#define HH    128
#define WW2   128
#define HW    16384
#define CHW   3145728          // 192*128*128
#define OUTSZ 25165824         // 8*CHW
#define NWIN  256
#define EPSN  1e-5f

#if defined(__has_builtin)
#if __has_builtin(__builtin_amdgcn_tensor_load_to_lds) && \
    __has_builtin(__builtin_amdgcn_s_wait_tensorcnt)
#define HAVE_TDM 1
#endif
#endif
#ifndef HAVE_TDM
#define HAVE_TDM 0
#endif

enum { EPI_GATE = 0, EPI_QKV = 1, EPI_OUT = 2, EPI_GELU = 3, EPI_ADD = 4 };

union FragU { v16bf v; unsigned u[8]; };
union Pk8   { uint4 q; bf16_t h[8]; };
union Pk8f  { float4 f[2]; float s[8]; };

// ---------- helpers ----------
__device__ __forceinline__ void tok2hw(int t, int& hh, int& ww) {
  int win = t >> 6, pos = t & 63;
  hh = ((win >> 4) << 3) | (pos >> 3);
  ww = ((win & 15) << 3) | (pos & 7);
}

// A fragment: 16x32 bf16 (MxK) row-major in LDS, per ISA wave32 layout.
// B fragment: 32x16 (KxN) staged N-major ([n][k]) so indexing is identical.
__device__ __forceinline__ v16bf load_frag(const bf16_t* base, int ld) {
  int lane = threadIdx.x & 31;
  int half = lane >> 4;
  int mn   = lane & 15;
  const unsigned* row = (const unsigned*)(base + mn * ld);
  FragU f;
#pragma unroll
  for (int r = 0; r < 8; ++r) {
    int kb = ((r >> 2) << 4) + half * 8 + ((r & 3) << 1);  // K pair start
    f.u[r] = row[kb >> 1];
  }
  return f.v;
}

__device__ __forceinline__ v8f wmma_bf16(v16bf a, v16bf b, v8f c) {
  return __builtin_amdgcn_wmma_f32_16x16x32_bf16(false, a, false, b, (short)0, c,
                                                 false, false);
}

#if HAVE_TDM
// Tensor Data Mover: DMA a 64x64-element (2-byte) 2D tile, row stride 256
// elements, from global memory into LDS. D# layout per CDNA5 ISA ch.8.
__device__ __forceinline__ void tdm_load_tile_2d(unsigned lds_addr,
                                                 unsigned long long gaddr) {
  u32x4 g0;
  g0[0] = 1u;                                            // count=1, user D#
  g0[1] = lds_addr;                                      // LDS byte address
  g0[2] = (unsigned)(gaddr & 0xFFFFFFFFull);             // global_addr[31:0]
  g0[3] = ((unsigned)(gaddr >> 32) & 0x1FFFFFFu) | (2u << 30);  // [56:32]|type=2
  i32x8 g1;
  g1[0] = 0x00010000;          // workgroup_mask=0, data_size=1 (2 bytes)
  g1[1] = (int)(256u << 16);   // tensor_dim0 = 256 elements (low16)
  g1[2] = (int)(16384u << 16); // tensor_dim1 = 16384 rows (low16)
  g1[3] = (int)(64u << 16);    // tile_dim0 = 64 elements
  g1[4] = 64;                  // tile_dim1 = 64 rows, tile_dim2 = 0
  g1[5] = 256;                 // tensor_dim0_stride = 256 elements
  g1[6] = 0;
  g1[7] = 0;
  i32x4 zz = {0, 0, 0, 0};
#if __clang_major__ >= 23
  i32x8 z8 = {0, 0, 0, 0, 0, 0, 0, 0};
  __builtin_amdgcn_tensor_load_to_lds(g0, g1, zz, zz, z8, 0);
#else
  __builtin_amdgcn_tensor_load_to_lds(g0, g1, zz, zz, 0);
#endif
}
#endif

// ---------- GroupNorm stats ----------
__global__ __launch_bounds__(256) void stats_kernel(const float* __restrict__ x0,
                                                    const float* __restrict__ x1,
                                                    float* __restrict__ out) {
  const float* x = blockIdx.y ? x1 : x0;
  __shared__ float ss[256], sq[256];
  int tid = threadIdx.x;
  const int chunk = CHW / 128;
  size_t begin = (size_t)blockIdx.x * chunk;
  float s = 0.f, q = 0.f;
  for (size_t i = begin + tid; i < begin + chunk; i += 256) {
    float v = x[i];
    s += v; q += v * v;
  }
  ss[tid] = s; sq[tid] = q;
  __syncthreads();
  for (int st = 128; st > 0; st >>= 1) {
    if (tid < st) { ss[tid] += ss[tid + st]; sq[tid] += sq[tid + st]; }
    __syncthreads();
  }
  if (tid == 0) {
    atomicAdd(&out[blockIdx.y * 2 + 0], ss[0]);
    atomicAdd(&out[blockIdx.y * 2 + 1], sq[0]);
  }
}

__global__ void finalize_kernel(const float* __restrict__ sums, float* __restrict__ mr) {
  if (threadIdx.x == 0) {
    for (int z = 0; z < 2; ++z) {
      float mean = sums[z * 2] / (float)CHW;
      float var  = sums[z * 2 + 1] / (float)CHW - mean * mean;
      mr[z * 2 + 0] = mean;
      mr[z * 2 + 1] = rsqrtf(var + EPSN);
    }
  }
}

// ---------- normalize BCHW -> window-major token bf16 ----------
__global__ __launch_bounds__(192) void normalize_kernel(
    const float* __restrict__ s0, const float* __restrict__ s1,
    const float* __restrict__ mr,
    const float* __restrict__ g0, const float* __restrict__ b0,
    const float* __restrict__ g1, const float* __restrict__ b1,
    bf16_t* __restrict__ d0, bf16_t* __restrict__ d1) {
  int z = blockIdx.z;
  const float* src = z ? s1 : s0;
  const float* gam = z ? g1 : g0;
  const float* bet = z ? b1 : b0;
  bf16_t* dst = z ? d1 : d0;
  float mean = mr[z * 2], rstd = mr[z * 2 + 1];
  int t = blockIdx.x, c = threadIdx.x;
  int hh, ww; tok2hw(t, hh, ww);
  float x = src[(size_t)c * HW + hh * WW2 + ww];
  dst[(size_t)t * 192 + c] = (bf16_t)((x - mean) * rstd * gam[c] + bet[c]);
}

// ---------- generic 64x192 tile GEMM (K multiple of 32) ----------
struct GArgs {
  const bf16_t* A0; const bf16_t* A1;      // bf16 activation (token-major, stride K)
  const float*  Af0; const float* Af1;     // fp32 activation (for EPI_OUT)
  const float*  W0; const float* W1;       // weights, row-major [O][K] fp32
  const float*  B0; const float* B1;       // bias
  const bf16_t* base0; const bf16_t* base1;// gate residual base
  bf16_t* db0; bf16_t* db1;                // bf16 dst
  float*  df0; float*  df1;                // fp32 dst
  const float* r0; const float* r1;        // fp32 residual src (BCHW)
  const float* sc0; const float* sc1;      // alpha / beta (device scalars)
};

template <int EPI, int KD, bool AF32>
__global__ __launch_bounds__(256) void gemm_kernel(GArgs g) {
  __shared__ __align__(16) bf16_t As[64 * 32];
  __shared__ __align__(16) bf16_t Bs[192 * 32];
  const int z  = blockIdx.z;
  const int yb = blockIdx.y;
  const int t0 = blockIdx.x * 64;
  const int tid = threadIdx.x;

  const bf16_t* Ab = z ? g.A1 : g.A0;
  const float*  Af = z ? g.Af1 : g.Af0;
  const float*  Wp = z ? g.W1 : g.W0;
  const float*  Bias = z ? g.B1 : g.B0;
  const int wRowOff = (EPI == EPI_QKV || EPI == EPI_GELU) ? yb * 192 : 0;

  const int wv = tid >> 5;
  const int lane = tid & 31;
  const int half = lane >> 4;
  const int cn = lane & 15;
  const int m_tile = wv >> 1;
  const int nbase = (wv & 1) * 96;

  v8f acc[6] = {};

  for (int k0 = 0; k0 < KD; k0 += 32) {
    // stage A tile: 64x32 (row-major)
    {
      int e = tid * 8;
      int row = e >> 5, col = e & 31;
      if (AF32) {
        Pk8f s; Pk8 d;
        const float* sp = Af + (size_t)(t0 + row) * KD + k0 + col;
        s.f[0] = *(const float4*)sp;
        s.f[1] = *(const float4*)(sp + 4);
#pragma unroll
        for (int q = 0; q < 8; ++q) d.h[q] = (bf16_t)s.s[q];
        *(uint4*)&As[e] = d.q;
        if (k0 + 32 < KD) __builtin_prefetch(sp + 32);
      } else {
        const bf16_t* sp = Ab + (size_t)(t0 + row) * KD + k0 + col;
        *(uint4*)&As[e] = *(const uint4*)sp;
        if (k0 + 32 < KD) __builtin_prefetch(sp + 32);
      }
    }
    // stage B tile: weights as N-major [192][32], fp32 -> bf16
#pragma unroll
    for (int gi = 0; gi < 3; ++gi) {
      int e = tid * 8 + gi * 2048;
      int row = e >> 5, col = e & 31;
      const float* sp = Wp + (size_t)(wRowOff + row) * KD + k0 + col;
      Pk8f s; Pk8 d;
      s.f[0] = *(const float4*)sp;
      s.f[1] = *(const float4*)(sp + 4);
#pragma unroll
      for (int q = 0; q < 8; ++q) d.h[q] = (bf16_t)s.s[q];
      *(uint4*)&Bs[e] = d.q;
      if (gi == 0 && k0 + 32 < KD) __builtin_prefetch(sp + 32);
    }
    __syncthreads();
    v16bf a = load_frag(As + m_tile * 16 * 32, 32);
#pragma unroll
    for (int j = 0; j < 6; ++j) {
      v16bf b = load_frag(Bs + (nbase + j * 16) * 32, 32);
      acc[j] = wmma_bf16(a, b, acc[j]);
    }
    __syncthreads();
  }

  // ---- epilogue ----
#pragma unroll
  for (int j = 0; j < 6; ++j) {
    int o = nbase + j * 16 + cn;
#pragma unroll
    for (int r = 0; r < 8; ++r) {
      int m = m_tile * 16 + r + half * 8;
      int t = t0 + m;
      float v = acc[j][r];
      if (EPI == EPI_GATE) {
        const bf16_t* base = z ? g.base1 : g.base0;
        bf16_t* dst = z ? g.db1 : g.db0;
        float sc = *(z ? g.sc1 : g.sc0);
        float x = v + Bias[o];
        float sg = 1.f / (1.f + __expf(-x));
        dst[(size_t)t * 192 + o] = (bf16_t)((float)base[(size_t)t * 192 + o] + sc * sg);
      } else if (EPI == EPI_QKV) {
        bf16_t* dst = (z ? g.db1 : g.db0) + (size_t)yb * TS * 256;
        float x = v + Bias[yb * 192 + o];
        int hh = o / 48, dd = o % 48;
        dst[(size_t)t * 256 + hh * 64 + dd] = (bf16_t)x;
      } else if (EPI == EPI_OUT) {
        float* dst = z ? g.df1 : g.df0;
        const float* res = z ? g.r1 : g.r0;
        int hh, ww; tok2hw(t, hh, ww);
        size_t idx = (size_t)o * HW + hh * WW2 + ww;
        dst[idx] = v + Bias[o] + res[idx];
      } else if (EPI == EPI_GELU) {
        bf16_t* dst = z ? g.db1 : g.db0;
        float x = v + Bias[yb * 192 + o];
        float ge = 0.5f * x * (1.f + erff(x * 0.70710678f));
        dst[(size_t)t * 384 + yb * 192 + o] = (bf16_t)ge;
      } else {  // EPI_ADD
        float* dst = z ? g.df1 : g.df0;
        int hh, ww; tok2hw(t, hh, ww);
        size_t idx = (size_t)o * HW + hh * WW2 + ww;
        dst[idx] += v + Bias[o];
      }
    }
  }
}

// ---------- windowed attention: one block per (window, head, stream) ----------
__global__ __launch_bounds__(128) void attn_kernel(const bf16_t* __restrict__ qkv0,
                                                   const bf16_t* __restrict__ qkv1,
                                                   const float* __restrict__ relb,
                                                   float* __restrict__ out0,
                                                   float* __restrict__ out1) {
  __shared__ __align__(16) bf16_t Qs[64 * 64];
  __shared__ __align__(16) bf16_t Ks[64 * 64];
  __shared__ __align__(16) bf16_t VsT[64 * 64];
  __shared__ __align__(16) bf16_t Ps[64 * 64];
  __shared__ float Ss[64 * 64];
  int win = blockIdx.x, h = blockIdx.y, z = blockIdx.z;
  const bf16_t* qkv = z ? qkv1 : qkv0;
  float* outp = z ? out1 : out0;
  const bf16_t* Qg = qkv;
  const bf16_t* Kg = qkv + (size_t)TS * 256;
  const bf16_t* Vg = qkv + (size_t)2 * TS * 256;
  int tid = threadIdx.x;
  int t0 = win * 64;

#if HAVE_TDM
  // Tensor Data Mover: wave 0 DMAs the Q and K 64x64 tiles into LDS,
  // tracked with TENSORcnt; V is staged transposed by all threads.
  if (tid < 32) {
    tdm_load_tile_2d((unsigned)(uintptr_t)&Qs[0],
                     (unsigned long long)(uintptr_t)(Qg + (size_t)t0 * 256 + h * 64));
    tdm_load_tile_2d((unsigned)(uintptr_t)&Ks[0],
                     (unsigned long long)(uintptr_t)(Kg + (size_t)t0 * 256 + h * 64));
    __builtin_amdgcn_s_wait_tensorcnt(0);
  }
#pragma unroll
  for (int gi = 0; gi < 4; ++gi) {
    int e = gi * 1024 + tid * 8;
    int row = e >> 6, col = e & 63;
    size_t src = (size_t)(t0 + row) * 256 + h * 64 + col;
    Pk8 pv; pv.q = *(const uint4*)&Vg[src];
#pragma unroll
    for (int q = 0; q < 8; ++q) VsT[(col + q) * 64 + row] = pv.h[q];
  }
#else
  // stage: Q/K row-major [tok][64], V transposed [d][tok]
#pragma unroll
  for (int gi = 0; gi < 4; ++gi) {
    int e = gi * 1024 + tid * 8;
    int row = e >> 6, col = e & 63;
    size_t src = (size_t)(t0 + row) * 256 + h * 64 + col;
    *(uint4*)&Qs[e] = *(const uint4*)&Qg[src];
    *(uint4*)&Ks[e] = *(const uint4*)&Kg[src];
    Pk8 pv; pv.q = *(const uint4*)&Vg[src];
#pragma unroll
    for (int q = 0; q < 8; ++q) VsT[(col + q) * 64 + row] = pv.h[q];
  }
#endif
  __syncthreads();

  int wv = tid >> 5, lane = tid & 31, half = lane >> 4, cn = lane & 15;

  // S = Q @ K^T  (64x64, K=64 padded; padding lanes are zero)
  v8f sacc[4] = {};
#pragma unroll
  for (int k0 = 0; k0 < 64; k0 += 32) {
    v16bf a = load_frag(Qs + wv * 16 * 64 + k0, 64);
#pragma unroll
    for (int j = 0; j < 4; ++j) {
      v16bf b = load_frag(Ks + j * 16 * 64 + k0, 64);
      sacc[j] = wmma_bf16(a, b, sacc[j]);
    }
  }
  const float scale = 0.14433756729740643f;  // 48^-0.5
  const float* rb = relb + h * 225;
#pragma unroll
  for (int j = 0; j < 4; ++j) {
    int kcol = j * 16 + cn;
    int ki = kcol >> 3, kj = kcol & 7;
#pragma unroll
    for (int r = 0; r < 8; ++r) {
      int qrow = wv * 16 + r + half * 8;
      int qi = qrow >> 3, qj = qrow & 7;
      int ridx = (qi - ki + 7) * 15 + (qj - kj + 7);
      Ss[qrow * 64 + kcol] = sacc[j][r] * scale + rb[ridx];
    }
  }
  __syncthreads();

  // row softmax -> P (bf16)
  if (tid < 64) {
    float* rowp = &Ss[tid * 64];
    float mx = rowp[0];
    for (int i = 1; i < 64; ++i) mx = fmaxf(mx, rowp[i]);
    float sum = 0.f;
    for (int i = 0; i < 64; ++i) {
      float e = __expf(rowp[i] - mx);
      rowp[i] = e; sum += e;
    }
    float inv = 1.f / sum;
    for (int i = 0; i < 64; ++i) Ps[tid * 64 + i] = (bf16_t)(rowp[i] * inv);
  }
  __syncthreads();

  // O = P @ V  (64x48 valid columns)
  v8f oacc[3] = {};
#pragma unroll
  for (int k0 = 0; k0 < 64; k0 += 32) {
    v16bf a = load_frag(Ps + wv * 16 * 64 + k0, 64);
#pragma unroll
    for (int j = 0; j < 3; ++j) {
      v16bf b = load_frag(VsT + j * 16 * 64 + k0, 64);
      oacc[j] = wmma_bf16(a, b, oacc[j]);
    }
  }
#pragma unroll
  for (int j = 0; j < 3; ++j) {
    int d = j * 16 + cn;
#pragma unroll
    for (int r = 0; r < 8; ++r) {
      int q = wv * 16 + r + half * 8;
      outp[(size_t)(t0 + q) * 192 + h * 48 + d] = oacc[j][r];
    }
  }
}

// ---------- launch ----------
extern "C" void kernel_launch(void* const* d_in, const int* in_sizes, int n_in,
                              void* d_out, int out_size, void* d_ws, size_t ws_size,
                              hipStream_t stream) {
  const float* sem  = (const float*)d_in[0];
  const float* dep  = (const float*)d_in[1];
  const float* ns_g = (const float*)d_in[2];  const float* ns_b = (const float*)d_in[3];
  const float* nd_g = (const float*)d_in[4];  const float* nd_b = (const float*)d_in[5];
  const float* wd2s = (const float*)d_in[6];  const float* bd2s = (const float*)d_in[7];
  const float* ws2d = (const float*)d_in[8];  const float* bs2d = (const float*)d_in[9];
  const float* alpha = (const float*)d_in[10]; const float* beta = (const float*)d_in[11];
  const float* qsw = (const float*)d_in[12]; const float* qsb = (const float*)d_in[13];
  const float* osw = (const float*)d_in[14]; const float* osb = (const float*)d_in[15];
  const float* qdw = (const float*)d_in[16]; const float* qdb = (const float*)d_in[17];
  const float* odw = (const float*)d_in[18]; const float* odb = (const float*)d_in[19];
  const float* relb = (const float*)d_in[20];
  const float* fs_g = (const float*)d_in[21]; const float* fs_b = (const float*)d_in[22];
  const float* fs_w1 = (const float*)d_in[23]; const float* fs_b1 = (const float*)d_in[24];
  const float* fs_w2 = (const float*)d_in[25]; const float* fs_b2 = (const float*)d_in[26];
  const float* fd_g = (const float*)d_in[27]; const float* fd_b = (const float*)d_in[28];
  const float* fd_w1 = (const float*)d_in[29]; const float* fd_b1 = (const float*)d_in[30];
  const float* fd_w2 = (const float*)d_in[31]; const float* fd_b2 = (const float*)d_in[32];
  float* dout = (float*)d_out;

  char* wsp = (char*)d_ws;
  size_t off = 0;
  float* stats = (float*)(wsp + off); off += 256;
  bf16_t* semN  = (bf16_t*)(wsp + off); off += (size_t)TS * 192 * 2;
  bf16_t* depN  = (bf16_t*)(wsp + off); off += (size_t)TS * 192 * 2;
  bf16_t* semIn = (bf16_t*)(wsp + off); off += (size_t)TS * 192 * 2;
  bf16_t* depIn = (bf16_t*)(wsp + off); off += (size_t)TS * 192 * 2;
  bf16_t* qkvS  = (bf16_t*)(wsp + off); off += (size_t)3 * TS * 256 * 2;
  bf16_t* qkvD  = (bf16_t*)(wsp + off); off += (size_t)3 * TS * 256 * 2;
  float*  attnS = (float*)(wsp + off); off += (size_t)TS * 192 * 4;
  float*  attnD = (float*)(wsp + off); off += (size_t)TS * 192 * 4;
  bf16_t* h1S = qkvS;                      // reuse qkv region after attention
  bf16_t* h1D = qkvS + (size_t)TS * 384;
  if (ws_size < off) return;

  for (int b = 0; b < 8; ++b) {
    const float* semB = sem + (size_t)b * CHW;
    const float* depB = dep + (size_t)b * CHW;
    float* outS = dout + (size_t)b * CHW;
    float* outD = dout + (size_t)OUTSZ + (size_t)b * CHW;

    // GroupNorm(1) both streams
    hipMemsetAsync(stats, 0, 64, stream);
    stats_kernel<<<dim3(128, 2), 256, 0, stream>>>(semB, depB, stats);
    finalize_kernel<<<1, 32, 0, stream>>>(stats, stats + 4);
    normalize_kernel<<<dim3(TS, 1, 2), 192, 0, stream>>>(semB, depB, stats + 4,
        ns_g, ns_b, nd_g, nd_b, semN, depN);

    // cross-gating: sem_in = sem_n + alpha*sigmoid(Wd2s . dep_n), and symmetric
    {
      GArgs g{};
      g.A0 = depN; g.A1 = semN; g.W0 = wd2s; g.W1 = ws2d; g.B0 = bd2s; g.B1 = bs2d;
      g.base0 = semN; g.base1 = depN; g.db0 = semIn; g.db1 = depIn;
      g.sc0 = alpha; g.sc1 = beta;
      gemm_kernel<EPI_GATE, 192, false><<<dim3(256, 1, 2), 256, 0, stream>>>(g);
    }

    // QKV projection into head-padded bf16 layout (pad dims zeroed)
    hipMemsetAsync(qkvS, 0, (size_t)2 * 3 * TS * 256 * 2, stream);
    {
      GArgs g{};
      g.A0 = semIn; g.A1 = depIn; g.W0 = qsw; g.W1 = qdw; g.B0 = qsb; g.B1 = qdb;
      g.db0 = qkvS; g.db1 = qkvD;
      gemm_kernel<EPI_QKV, 192, false><<<dim3(256, 3, 2), 256, 0, stream>>>(g);
    }

    // windowed attention
    attn_kernel<<<dim3(NWIN, 4, 2), 128, 0, stream>>>(qkvS, qkvD, relb, attnS, attnD);

    // output projection + residual with original input, write d_out
    {
      GArgs g{};
      g.Af0 = attnS; g.Af1 = attnD; g.W0 = osw; g.W1 = odw; g.B0 = osb; g.B1 = odb;
      g.r0 = semB; g.r1 = depB; g.df0 = outS; g.df1 = outD;
      gemm_kernel<EPI_OUT, 192, true><<<dim3(256, 1, 2), 256, 0, stream>>>(g);
    }

    // FFN: GroupNorm -> D->2D GELU -> 2D->D, accumulate into d_out
    hipMemsetAsync((char*)stats + 32, 0, 16, stream);
    stats_kernel<<<dim3(128, 2), 256, 0, stream>>>(outS, outD, stats + 8);
    finalize_kernel<<<1, 32, 0, stream>>>(stats + 8, stats + 12);
    normalize_kernel<<<dim3(TS, 1, 2), 192, 0, stream>>>(outS, outD, stats + 12,
        fs_g, fs_b, fd_g, fd_b, semN, depN);
    {
      GArgs g{};
      g.A0 = semN; g.A1 = depN; g.W0 = fs_w1; g.W1 = fd_w1; g.B0 = fs_b1; g.B1 = fd_b1;
      g.db0 = h1S; g.db1 = h1D;
      gemm_kernel<EPI_GELU, 192, false><<<dim3(256, 2, 2), 256, 0, stream>>>(g);
    }
    {
      GArgs g{};
      g.A0 = h1S; g.A1 = h1D; g.W0 = fs_w2; g.W1 = fd_w2; g.B0 = fs_b2; g.B1 = fd_b2;
      g.df0 = outS; g.df1 = outD;
      gemm_kernel<EPI_ADD, 384, false><<<dim3(256, 1, 2), 256, 0, stream>>>(g);
    }
  }
}
